// DCNv3Layer_61564061221317
// MI455X (gfx1250) — matrix operand order repaired and checked by reading, not compiled, not run on hardware
//
#include <hip/hip_runtime.h>
#include <hip/hip_bf16.h>
#include <stdint.h>

// ---------- types for WMMA fragments ----------
typedef __attribute__((ext_vector_type(16))) __bf16 v16bf;
typedef __attribute__((ext_vector_type(8)))  float  v8f;
typedef __attribute__((ext_vector_type(4)))  unsigned int u32x4;

union FragBF {
  v16bf v;
  u32x4 q[2];
};

__device__ __forceinline__ uint16_t f2bf(float f) {
  uint32_t u = __float_as_uint(f);
  uint32_t r = u + 0x7FFFu + ((u >> 16) & 1u);
  return (uint16_t)(r >> 16);
}

// ---------- pack weight (K x N row-major f32) into WMMA B-fragment layout ----------
// layout: [kb = k/32][nt = n/16][lane = 16*((k%32)/16) + (n%16)][e = k%16] (bf16)
__global__ void pack_b_kernel(const float* __restrict__ W, uint16_t* __restrict__ out,
                              int K, int N) {
  int tid = blockIdx.x * blockDim.x + threadIdx.x;
  if (tid >= K * N) return;
  int k = tid / N, n = tid - k * N;
  int kb = k >> 5, kr = k & 31;
  int hi = kr >> 4, e = kr & 15;
  int lane = hi * 16 + (n & 15);
  int nt = n >> 4;
  size_t idx = ((size_t)(kb * (N >> 4) + nt) * 32 + lane) * 16 + e;
  out[idx] = f2bf(W[(size_t)k * N + n]);
}

// ---------- x: NCHW f32 -> NHWC bf16 ----------
__global__ void x_nhwc_bf16_kernel(const float* __restrict__ x, uint16_t* __restrict__ xb,
                                   int M) {
  int tid = blockIdx.x * blockDim.x + threadIdx.x;
  if (tid >= M * 256) return;
  int m = tid >> 8, c = tid & 255;
  int img = m >> 12, hw = m & 4095;
  xb[tid] = f2bf(x[(((size_t)img * 256 + c) << 12) + hw]);
}

// ---------- depthwise 3x3 conv + LayerNorm + GELU(erf) -> NHWC bf16 ----------
// one wave per pixel, each lane owns 8 channels; wave-shuffle reduction for mean/var
__global__ __launch_bounds__(256) void dw_ln_gelu_kernel(
    const float* __restrict__ x, const float* __restrict__ dw_w,
    const float* __restrict__ dw_b, const float* __restrict__ ln_g,
    const float* __restrict__ ln_b, uint16_t* __restrict__ x1b, int M) {
  int wave = blockIdx.x * 8 + (threadIdx.x >> 5);
  if (wave >= M) return;
  int lane = threadIdx.x & 31;
  int img = wave >> 12, hw = wave & 4095;
  int h = hw >> 6, w = hw & 63;
  float v[8], s = 0.f, ss = 0.f;
#pragma unroll
  for (int j = 0; j < 8; ++j) {
    int c = j * 32 + lane;
    float acc = dw_b[c];
    const float* xp = x + (((size_t)img * 256 + c) << 12);
    const float* wp = dw_w + c * 9;
#pragma unroll
    for (int ky = 0; ky < 3; ++ky) {
      int hh = h + ky - 1;
      if (hh < 0 || hh >= 64) continue;
#pragma unroll
      for (int kx = 0; kx < 3; ++kx) {
        int ww = w + kx - 1;
        if (ww < 0 || ww >= 64) continue;
        acc += xp[hh * 64 + ww] * wp[ky * 3 + kx];
      }
    }
    v[j] = acc; s += acc; ss += acc * acc;
  }
#pragma unroll
  for (int o = 16; o > 0; o >>= 1) {
    s  += __shfl_xor(s,  o, 32);
    ss += __shfl_xor(ss, o, 32);
  }
  float mu  = s * (1.f / 256.f);
  float var = ss * (1.f / 256.f) - mu * mu;
  float rinv = rsqrtf(var + 1e-5f);
#pragma unroll
  for (int j = 0; j < 8; ++j) {
    int c = j * 32 + lane;
    float t = (v[j] - mu) * rinv * ln_g[c] + ln_b[c];
    float ge = 0.5f * t * (1.f + erff(t * 0.70710678118f));
    x1b[(size_t)wave * 256 + c] = f2bf(ge);
  }
}

// ---------- WMMA GEMM: D[M x N] = A_bf16[M x 256] @ Bpacked + bias ----------
// each wave computes a 16 x (16*NT) strip: NT independent accumulators share one
// A fragment per k-step -> 2 + 2*NT b128 loads feed NT back-to-back WMMAs,
// breaking the WMMA->WMMA RAW chain without v_nop padding.
template <int NT>
__global__ __launch_bounds__(256) void gemm_wmma_bf16(
    const uint16_t* __restrict__ A, const uint16_t* __restrict__ Bp,
    const float* __restrict__ bias, float* __restrict__ D,
    int Mt, int NtTot, int store_nchw) {
  const int nGroups = NtTot / NT;
  int wave = blockIdx.x * 8 + (threadIdx.x >> 5);
  if (wave >= Mt * nGroups) return;
  int mt = wave / nGroups, ng = wave - mt * nGroups;
  int nt0 = ng * NT;
  int lane = threadIdx.x & 31;
  int hi = lane >> 4, l16 = lane & 15;
  const int K = 256;
  int N = NtTot << 4;
  // A fragment: lane (hi,l16) holds row m=l16, K = {16hi..16hi+15} per 32-block
  const uint16_t* arow  = A + (size_t)(mt * 16 + l16) * K + 16 * hi;
  const uint16_t* bbase = Bp + ((size_t)nt0 * 32 + lane) * 16;
  const size_t bstride  = (size_t)NtTot * 32 * 16;  // elements per k-block
  v8f acc[NT];
#pragma unroll
  for (int t = 0; t < NT; ++t) acc[t] = {};
#pragma unroll
  for (int kb = 0; kb < K / 32; ++kb) {
    FragBF a;
    const uint16_t* ap = arow + kb * 32;
    a.q[0] = *(const u32x4*)(ap);
    a.q[1] = *(const u32x4*)(ap + 8);
    const uint16_t* bp = bbase + (size_t)kb * bstride;
#pragma unroll
    for (int t = 0; t < NT; ++t) {
      FragBF b;
      b.q[0] = *(const u32x4*)(bp + t * 512);      // 512 = 32 lanes * 16 elems per n-tile
      b.q[1] = *(const u32x4*)(bp + t * 512 + 8);
      acc[t] = __builtin_amdgcn_wmma_f32_16x16x32_bf16(
          false, a.v, false, b.v, (short)0, acc[t], false, false);
    }
  }
#pragma unroll
  for (int t = 0; t < NT; ++t) {
    int n = (nt0 + t) * 16 + l16;
    float bv = bias[n];
#pragma unroll
    for (int r = 0; r < 8; ++r) {
      int m = mt * 16 + r + 8 * hi;
      float val = acc[t][r] + bv;
      if (!store_nchw) {
        D[(size_t)m * N + n] = val;
      } else {  // NCHW scatter: C==N==256, HW==4096
        int img = m >> 12, hw = m & 4095;
        D[(((size_t)img * 256 + n) << 12) + hw] = val;
      }
    }
  }
}

// ---------- softmax over P=9 per (pixel, group) ----------
__global__ void softmax9_kernel(float* __restrict__ mask, int M) {
  int tid = blockIdx.x * blockDim.x + threadIdx.x;
  if (tid >= M * 16) return;
  float* p = mask + (size_t)(tid >> 4) * 144 + (tid & 15) * 9;
  float v[9], mx = -1e30f;
#pragma unroll
  for (int i = 0; i < 9; ++i) { v[i] = p[i]; mx = fmaxf(mx, v[i]); }
  float s = 0.f;
#pragma unroll
  for (int i = 0; i < 9; ++i) { v[i] = __expf(v[i] - mx); s += v[i]; }
  float r = 1.f / s;
#pragma unroll
  for (int i = 0; i < 9; ++i) p[i] = v[i] * r;
}

// ---------- deformable sampling core -> y NHWC bf16 ----------
// reference math collapses to padded pixel coords: px = w+1+kx+offx, py = h+1+ky+offy
__global__ void dcnv3_kernel(const float* __restrict__ xq, const float* __restrict__ offs,
                             const float* __restrict__ mask, uint16_t* __restrict__ yb,
                             int M) {
  int tid = blockIdx.x * blockDim.x + threadIdx.x;
  if (tid >= M * 256) return;
  int m = tid >> 8, c = tid & 255;
  int g = c >> 4;
  int img = m >> 12, hw = m & 4095;
  int h = hw >> 6, w = hw & 63;
  const float* op = offs + (size_t)m * 288 + g * 18;
  const float* mp = mask + (size_t)m * 144 + g * 9;
  const float* xbase = xq + (((size_t)img) << 12) * 256 + c;
  float acc = 0.f;
#pragma unroll
  for (int p = 0; p < 9; ++p) {
    int kx = p / 3 - 1;  // kw-major flatten: gx over w
    int ky = p % 3 - 1;
    float px = (float)(w + 1 + kx) + op[p * 2 + 0];
    float py = (float)(h + 1 + ky) + op[p * 2 + 1];
    float x0f = floorf(px), y0f = floorf(py);
    float fx = px - x0f, fy = py - y0f;
    int x0 = (int)x0f, y0 = (int)y0f;
    float sm = 0.f;
    auto tap = [&](int ix, int iy, float wt) {
      // contribution nonzero only inside the un-padded interior (pad ring is zeros)
      if ((unsigned)(ix - 1) < 64u && (unsigned)(iy - 1) < 64u)
        sm += wt * xbase[(size_t)(((iy - 1) << 6) + (ix - 1)) * 256];
    };
    tap(x0,     y0,     (1.f - fx) * (1.f - fy));
    tap(x0 + 1, y0,     fx         * (1.f - fy));
    tap(x0,     y0 + 1, (1.f - fx) * fy);
    tap(x0 + 1, y0 + 1, fx         * fy);
    acc += mp[p] * sm;
  }
  yb[tid] = f2bf(acc);
}

extern "C" void kernel_launch(void* const* d_in, const int* in_sizes, int n_in,
                              void* d_out, int out_size, void* d_ws, size_t ws_size,
                              hipStream_t stream) {
  const float* x      = (const float*)d_in[0];
  const float* w_in   = (const float*)d_in[1];
  const float* b_in   = (const float*)d_in[2];
  const float* dw_w   = (const float*)d_in[3];
  const float* dw_b   = (const float*)d_in[4];
  const float* ln_g   = (const float*)d_in[5];
  const float* ln_b   = (const float*)d_in[6];
  const float* w_off  = (const float*)d_in[7];
  const float* b_off  = (const float*)d_in[8];
  const float* w_mask = (const float*)d_in[9];
  const float* b_mask = (const float*)d_in[10];
  const float* w_out  = (const float*)d_in[11];
  const float* b_out  = (const float*)d_in[12];
  float* out = (float*)d_out;

  const int M = 16384;  // N*H*W = 4*64*64
  char* ws = (char*)d_ws;
  uint16_t* Xb    = (uint16_t*)(ws + 0);          //  8 MB: x NHWC bf16
  uint16_t* X1b   = (uint16_t*)(ws + 8388608);    //  8 MB: branch feats bf16
  float*    Xq    = (float*)   (ws + 16777216);   // 16 MB: value feats f32 NHWC
  float*    Offs  = (float*)   (ws + 33554432);   // 18.9 MB: offsets
  float*    Msk   = (float*)   (ws + 52428800);   //  9.4 MB: masks
  uint16_t* Yb    = (uint16_t*)(ws + 61865984);   //  8 MB: sampled out bf16
  uint16_t* PKin  = (uint16_t*)(ws + 70254592);   // packed weights (bf16)
  uint16_t* PKoff = (uint16_t*)(ws + 70385664);
  uint16_t* PKmsk = (uint16_t*)(ws + 70533120);
  uint16_t* PKout = (uint16_t*)(ws + 70606848);

  // 1) pack weights to WMMA B-fragment layout (bf16)
  pack_b_kernel<<<(256 * 256 + 255) / 256, 256, 0, stream>>>(w_in,  PKin,  256, 256);
  pack_b_kernel<<<(256 * 288 + 255) / 256, 256, 0, stream>>>(w_off, PKoff, 256, 288);
  pack_b_kernel<<<(256 * 144 + 255) / 256, 256, 0, stream>>>(w_mask,PKmsk, 256, 144);
  pack_b_kernel<<<(256 * 256 + 255) / 256, 256, 0, stream>>>(w_out, PKout, 256, 256);

  // 2) activation staging
  x_nhwc_bf16_kernel<<<M, 256, 0, stream>>>(x, Xb, M);
  dw_ln_gelu_kernel<<<M / 8, 256, 0, stream>>>(x, dw_w, dw_b, ln_g, ln_b, X1b, M);

  // 3) WMMA GEMMs: value / offset / mask projections
  // waves = Mt * (NtTot/NT); 8 waves per block
  gemm_wmma_bf16<4><<<((M / 16) * (16 / 4) + 7) / 8, 256, 0, stream>>>(
      Xb,  PKin,  b_in,   Xq,   M / 16, 16, 0);
  gemm_wmma_bf16<3><<<((M / 16) * (18 / 3) + 7) / 8, 256, 0, stream>>>(
      X1b, PKoff, b_off,  Offs, M / 16, 18, 0);
  gemm_wmma_bf16<3><<<((M / 16) * (9 / 3) + 7) / 8, 256, 0, stream>>>(
      X1b, PKmsk, b_mask, Msk,  M / 16, 9,  0);

  // 4) softmax + deformable bilinear sampling
  softmax9_kernel<<<(M * 16 + 255) / 256, 256, 0, stream>>>(Msk, M);
  dcnv3_kernel<<<M, 256, 0, stream>>>(Xq, Offs, Msk, Yb, M);

  // 5) output projection, scatter straight to NCHW f32
  gemm_wmma_bf16<4><<<((M / 16) * (16 / 4) + 7) / 8, 256, 0, stream>>>(
      Yb, PKout, b_out, out, M / 16, 16, 1);
}
